// HyperMoMixLinear_64321430225160
// MI455X (gfx1250) — matrix operand order repaired and checked by reading, not compiled, not run on hardware
//
#include <hip/hip_runtime.h>

typedef __attribute__((ext_vector_type(16))) __bf16 v16bf;
typedef __attribute__((ext_vector_type(8)))  __bf16 v8bf;
typedef __attribute__((ext_vector_type(8)))  float  v8f;

#define T_TOK   2048      // B*S
#define DIN     1024
#define DOUT    1024
#define CDIM    64
#define KE      160       // extended feature dim 130 padded to multiple of 32
#define KT      5         // KE/32 WMMA K-steps
#define NB      32        // N_IN == N_OUT
#define DFLAT   32768     // flat Monarch weight size per token
#define NTILES  (DFLAT/16)
#define FRAG_ELEMS 512    // one B fragment: 32 lanes * 16 bf16 (K32 x N16)
#define XS_STRIDE 36      // padded i-row stride (floats): 144B, 16B-aligned, low-conflict
#define XS_GSZ   (32*XS_STRIDE)      // 1152 floats per g-slice

#if __has_builtin(__builtin_amdgcn_global_load_async_to_lds_b32)
#define ATHENA_ASYNC_LDS 1
#else
#define ATHENA_ASYNC_LDS 0
#endif

typedef __attribute__((address_space(1))) int as1_int;
typedef __attribute__((address_space(3))) int as3_int;

static __device__ __forceinline__ unsigned short f2bf(float f) {
    unsigned int u = __float_as_uint(f);
    u += 0x7FFFu + ((u >> 16) & 1u);          // round-to-nearest-even
    return (unsigned short)(u >> 16);
}

// ---------------------------------------------------------------------------
// Repack W[k=2][64][32768] + b[2][32768] (f32) into bf16 WMMA-B fragments:
// fragment (nTile, kt) holds K=kt*32..+31, N=nTile*16..+15; per-lane 16
// contiguous bf16 (32B) so the GEMM loads are 2x global_load_b128 per lane.
// Row map: rows 0..63 = W[0], 64..127 = W[1], 128/129 = b[0]/b[1], rest 0.
// ---------------------------------------------------------------------------
__global__ void pack_w(const float* __restrict__ W, const float* __restrict__ b,
                       unsigned short* __restrict__ Wbf) {
    int idx  = blockIdx.x * 256 + threadIdx.x;    // over NTILES*KT*512
    int e    = idx & 15;
    int lane = (idx >> 4) & 31;
    int frag = idx >> 9;
    int kt    = frag % KT;
    int nTile = frag / KT;
    int kloc  = (e < 8) ? ((lane >> 4) * 8 + e) : (16 + (lane >> 4) * 8 + (e - 8));   // B 32x16 layout (16-bit, table order)
    int n     = nTile * 16 + (lane & 15);
    int kk    = kt * 32 + kloc;
    float v;
    if      (kk < 128)  v = W[(size_t)kk * DFLAT + n];   // k*64+c == kk
    else if (kk == 128) v = b[n];
    else if (kk == 129) v = b[(size_t)DFLAT + n];
    else                v = 0.0f;
    Wbf[idx] = f2bf(v);
}

// ---------------------------------------------------------------------------
// Per-token prep: cf = tanh(x@Wc+bc); coeffs = softmax(cf@Wm+bm);
// ratio = cf@Wr+br; bias = cf@Wb+bb; e = [c0*cf, c1*cf, c0, c1, 0-pad] (bf16)
// 4 tokens per 256-thread block (64 threads/token).
// ---------------------------------------------------------------------------
__global__ __launch_bounds__(256) void prep_kernel(
    const float* __restrict__ x,  const float* __restrict__ Wc, const float* __restrict__ bc,
    const float* __restrict__ Wm, const float* __restrict__ bm,
    const float* __restrict__ Wr, const float* __restrict__ br,
    const float* __restrict__ Wb, const float* __restrict__ bb,
    unsigned short* __restrict__ ebf, float* __restrict__ ratio,
    float* __restrict__ biasf)
{
    __shared__ float cf_s[4][CDIM];
    __shared__ float coef_s[4][2];
    int sub = threadIdx.x >> 6;
    int c   = threadIdx.x & 63;
    int tok = blockIdx.x * 4 + sub;

    const float* xr = x + (size_t)tok * DIN;
    float acc = bc[c];
    for (int j = 0; j < DIN; ++j) acc += xr[j] * Wc[j * CDIM + c];
    float cf = tanhf(acc);
    cf_s[sub][c] = cf;
    __syncthreads();

    if (c == 0) {
        float m0 = bm[0], m1 = bm[1], r = br[0];
        for (int cc = 0; cc < CDIM; ++cc) {
            float v = cf_s[sub][cc];
            m0 += v * Wm[cc * 2 + 0];
            m1 += v * Wm[cc * 2 + 1];
            r  += v * Wr[cc];
        }
        float mx = fmaxf(m0, m1);
        float e0 = __expf(m0 - mx), e1 = __expf(m1 - mx);
        float inv = 1.0f / (e0 + e1);
        coef_s[sub][0] = e0 * inv;
        coef_s[sub][1] = e1 * inv;
        ratio[tok] = r;
    }
    __syncthreads();

    float c0 = coef_s[sub][0], c1 = coef_s[sub][1];
    unsigned short* er = ebf + (size_t)tok * KE;
    er[c]        = f2bf(c0 * cf);
    er[CDIM + c] = f2bf(c1 * cf);
    if (c < 2)            er[128 + c] = f2bf(c == 0 ? c0 : c1);
    if (c >= 2 && c < 32) er[128 + c] = 0;     // pad rows 130..159

    for (int it = 0; it < DOUT / CDIM; ++it) {
        int o = it * CDIM + c;
        float a2 = bb[o];
        for (int cc = 0; cc < CDIM; ++cc) a2 += cf_s[sub][cc] * Wb[cc * DOUT + o];
        biasf[(size_t)tok * DOUT + o] = a2;
    }
}

// ---------------------------------------------------------------------------
// Fused generator GEMM + Monarch contraction.
//  Stage1: h1T[t,o,g]  = sum_i x[t,g*32+i]   * (e[t]@W1')[g*1024+i*32+o]
//  Stage2: out[t,g,o]  = (sum_i h1T[t,g,i]   * (e[t]@W2')[g*1024+i*32+o])
//                        * ratio[t] + bias[t,g*32+o]
// h1 is stored transposed-within-token so BOTH stages stage their per-token
// operand with the same coalesced (token, 32-contiguous-cols) pattern.
// 32-token M-tile: two A fragments share every B fragment (halves L2 traffic).
// Per-token operands staged in LDS (async global->LDS when available) and
// consumed as aligned ds_load_b128, keeping the hot loop's VMEM for B only.
// ---------------------------------------------------------------------------
template<bool STAGE2>
__global__ __launch_bounds__(256) void monarch_stage(
    const float* __restrict__ xsrc,            // stage1: x ; stage2: h1T
    const unsigned short* __restrict__ ebf_us,
    const unsigned short* __restrict__ wbf_us,
    float* __restrict__ outp,                  // stage1: h1T ; stage2: out
    const float* __restrict__ ratio,
    const float* __restrict__ biasf)
{
    __shared__ float xs[4 * XS_GSZ];           // [gi][i][36] (padded, 18KB)

    int tid  = threadIdx.x;
    int lane = tid & 31;
    int wave = tid >> 5;
    int hf   = lane >> 4;
    int lh   = lane & 15;
    int tokBase = blockIdx.x * 32;
    int gBase   = blockIdx.y * 4;
    int gi = wave >> 1;                        // this wave's g slice (0..3)
    int g  = gBase + gi;
    int oh = wave & 1;

    // ---- stage per-token operand tile into LDS: xs[gi][i][t] -------------
    // global side: 32 contiguous cols per (token, g) -> coalesced over i.
    #pragma unroll
    for (int it = 0; it < 16; ++it) {
        int l  = it * 256 + tid;               // 4096 elements
        int i  = l & 31;
        int t  = (l >> 5) & 31;
        int sg = l >> 10;
        const float* gp = xsrc + (size_t)(tokBase + t) * DIN + (gBase + sg) * NB + i;
        float* lp = &xs[sg * XS_GSZ + i * XS_STRIDE + t];
#if ATHENA_ASYNC_LDS
        __builtin_amdgcn_global_load_async_to_lds_b32(
            (as1_int*)gp, (as3_int*)lp, 0, 0);
#else
        *lp = *gp;
#endif
    }
#if ATHENA_ASYNC_LDS
#if __has_builtin(__builtin_amdgcn_s_wait_asynccnt)
    __builtin_amdgcn_s_wait_asynccnt(0);
#else
    asm volatile("s_wait_asynccnt 0" ::: "memory");
#endif
#endif
    __syncthreads();

    // ---- A fragments (16-bit A 16x32: lanes 0-15 K 0-7/16-23, 16-31 K 8-15/24-31)
    v16bf a0[KT], a1[KT];
    #pragma unroll
    for (int kt = 0; kt < KT; ++kt) {
        const unsigned short* er0 = ebf_us + (size_t)(tokBase + lh)      * KE + kt * 32 + hf * 8;
        const unsigned short* er1 = ebf_us + (size_t)(tokBase + 16 + lh) * KE + kt * 32 + hf * 8;
        v8bf lo0 = *(const v8bf*)er0;
        v8bf hi0 = *(const v8bf*)(er0 + 16);
        v8bf lo1 = *(const v8bf*)er1;
        v8bf hi1 = *(const v8bf*)(er1 + 16);
        v16bf av0, av1;
        #pragma unroll
        for (int j = 0; j < 8; ++j) {
            av0[j] = lo0[j]; av0[j + 8] = hi0[j];
            av1[j] = lo1[j]; av1[j + 8] = hi1[j];
        }
        a0[kt] = av0;
        a1[kt] = av1;
    }

    v8f hacc0 = {};
    v8f hacc1 = {};

    for (int i = 0; i < NB; ++i) {
        int nTile = g * 64 + i * 2 + oh;
        const __bf16* fb = (const __bf16*)wbf_us
                         + (size_t)nTile * KT * FRAG_ELEMS + lane * 16;
        v8f d0 = {};
        v8f d1 = {};
        #pragma unroll
        for (int kt = 0; kt < KT; ++kt) {
            v16bf bt = *(const v16bf*)(fb + kt * FRAG_ELEMS);
            d0 = __builtin_amdgcn_wmma_f32_16x16x32_bf16(false, a0[kt], false, bt,
                                                         (short)0, d0, false, false);
            d1 = __builtin_amdgcn_wmma_f32_16x16x32_bf16(false, a1[kt], false, bt,
                                                         (short)0, d1, false, false);
        }
        // per-token operand from LDS: 4x aligned ds_load_b128
        int xbase = gi * XS_GSZ + i * XS_STRIDE + hf * 8;
        float xv0[8], xv1[8];
        *(float4*)&xv0[0] = *(const float4*)&xs[xbase];
        *(float4*)&xv0[4] = *(const float4*)&xs[xbase + 4];
        *(float4*)&xv1[0] = *(const float4*)&xs[xbase + 16];
        *(float4*)&xv1[4] = *(const float4*)&xs[xbase + 20];
        #pragma unroll
        for (int v = 0; v < 8; ++v) {
            hacc0[v] += xv0[v] * d0[v];
            hacc1[v] += xv1[v] * d1[v];
        }
    }

    int o = oh * 16 + lh;                       // output col within this g block
    #pragma unroll
    for (int v = 0; v < 8; ++v) {
        int t = hf * 8 + v;
        if (STAGE2) {
            // out[t, g*32+o] = h2 * ratio + bias
            size_t o0 = (size_t)(tokBase + t)      * DOUT + g * NB + o;
            size_t o1 = (size_t)(tokBase + 16 + t) * DOUT + g * NB + o;
            outp[o0] = hacc0[v] * ratio[tokBase + t]      + biasf[o0];
            outp[o1] = hacc1[v] * ratio[tokBase + 16 + t] + biasf[o1];
        } else {
            // transposed within token: h1T[t, o, g]
            size_t o0 = (size_t)(tokBase + t)      * DOUT + o * NB + g;
            size_t o1 = (size_t)(tokBase + 16 + t) * DOUT + o * NB + g;
            outp[o0] = hacc0[v];
            outp[o1] = hacc1[v];
        }
    }
}

// ---------------------------------------------------------------------------
extern "C" void kernel_launch(void* const* d_in, const int* in_sizes, int n_in,
                              void* d_out, int out_size, void* d_ws, size_t ws_size,
                              hipStream_t stream) {
    const float* x  = (const float*)d_in[0];
    const float* Wc = (const float*)d_in[1];
    const float* bc = (const float*)d_in[2];
    const float* W1 = (const float*)d_in[3];
    const float* b1 = (const float*)d_in[4];
    const float* W2 = (const float*)d_in[5];
    const float* b2 = (const float*)d_in[6];
    const float* Wm = (const float*)d_in[7];
    const float* bm = (const float*)d_in[8];
    const float* Wb = (const float*)d_in[9];
    const float* bb = (const float*)d_in[10];
    const float* Wr = (const float*)d_in[11];
    const float* br = (const float*)d_in[12];
    float* out = (float*)d_out;

    char* ws = (char*)d_ws;
    size_t off = 0;
    auto walloc = [&](size_t bytes) -> void* {
        off = (off + 255) & ~(size_t)255;
        void* p = ws + off;
        off += bytes;
        return p;
    };
    const size_t packedBytes = (size_t)NTILES * KT * FRAG_ELEMS * 2;   // 10 MB each
    unsigned short* W1bf = (unsigned short*)walloc(packedBytes);
    unsigned short* W2bf = (unsigned short*)walloc(packedBytes);
    unsigned short* ebf  = (unsigned short*)walloc((size_t)T_TOK * KE * 2);
    float* ratio = (float*)walloc((size_t)T_TOK * 4);
    float* biasf = (float*)walloc((size_t)T_TOK * DOUT * 4);
    float* h1ws  = (float*)walloc((size_t)T_TOK * DOUT * 4);

    const int packElems = NTILES * KT * FRAG_ELEMS;          // 5,242,880
    pack_w<<<packElems / 256, 256, 0, stream>>>(W1, b1, W1bf);
    pack_w<<<packElems / 256, 256, 0, stream>>>(W2, b2, W2bf);

    prep_kernel<<<T_TOK / 4, 256, 0, stream>>>(x, Wc, bc, Wm, bm, Wr, br, Wb, bb,
                                               ebf, ratio, biasf);

    dim3 grid(T_TOK / 32, 8);
    monarch_stage<false><<<grid, 256, 0, stream>>>(x,    ebf, W1bf, h1ws, ratio, biasf);
    monarch_stage<true ><<<grid, 256, 0, stream>>>(h1ws, ebf, W2bf, out,  ratio, biasf);

    (void)in_sizes; (void)n_in; (void)out_size; (void)ws_size; (void)b2;
}
